// GATModel_61091614818615
// MI455X (gfx1250) — matrix operand order repaired and checked by reading, not compiled, not run on hardware
//
#include <hip/hip_runtime.h>
#include <math.h>

// Problem constants (from the reference)
#define IN_CH   128
#define HID     32
#define HEADS   4
#define HC      (HEADS * HID)   // 128
#define OUT_CH  16
#define NEG_SLOPE 0.2f

typedef __attribute__((ext_vector_type(16))) _Float16 v16h;
typedef __attribute__((ext_vector_type(8)))  float    v8f;
typedef __attribute__((ext_vector_type(4)))  unsigned int u32x4;
typedef __attribute__((ext_vector_type(8)))  int      i32x8;
typedef __attribute__((ext_vector_type(4)))  int      i32x4;

__device__ __forceinline__ float leakyf(float x) { return x > 0.f ? x : NEG_SLOPE * x; }

// Float atomic-max via sign trick (monotone int mapping); all inputs finite.
__device__ __forceinline__ void atomicMaxF(float* addr, float val) {
  if (val >= 0.f) atomicMax((int*)addr, __float_as_int(val));
  else            atomicMin((unsigned int*)addr, __float_as_uint(val));
}

// ---------------------------------------------------------------------------
// C[M x N] = A[M x K] @ B[K x N], row-major f32; f32->f16 conversion in
// registers, math via v_wmma_f32_16x16x32_f16.
//
// Block = 256 threads (8 waves). The whole B matrix (K x ldn, <= 64 KB f32)
// is staged into LDS ONCE per block by a single Tensor Data Mover descriptor
// (tensor == tile == K x ldn, data_size = 4B, dim0 stride = ldn), then each
// wave computes one 16x16 tile reading its B fragments from LDS. This trades
// redundant global B traffic for ds_load traffic — the shape CDNA5's
// 320 KB-per-WGP LDS rewards.
//
// Wave->tile mapping: CT = ldn/16 column tiles; wave w -> (rowtile =
// blockIdx.x*(8/CT) + w/CT, coltile = w%CT). Tail waves exit AFTER the
// staging barrier, so __syncthreads stays uniform and surviving waves keep
// EXEC all-ones for WMMA.
//
// Fragment layouts per CDNA5 ISA 7.12.2 (wave32):
//   A (16x32 f16): lane = g*16+m, reg r holds K = (r<4 ? 8g+2r : 16+8g+2(r-4)) (+1)
//   B (32x16 f16): lane = g*16+n, reg r holds K = (r<4 ? 8g+2r : 16+8g+2(r-4)) (+1)
//   C (16x16 f32): reg r, lane g*16+n -> (M = 8g+r, N = n)
// ---------------------------------------------------------------------------
__global__ void gemm_wmma_f16(const float* __restrict__ A,
                              const float* __restrict__ B,
                              float* __restrict__ C,
                              int K, int ldn /* = N = ldb = ldc */,
                              int mtiles) {
  extern __shared__ float ldsW[];  // K x ldn f32, row-major

  const int lane = threadIdx.x & 31;
  const int wave = threadIdx.x >> 5;          // 0..7
  const int g    = lane >> 4;                 // half-wave group
  const int mn   = lane & 15;                 // A row / B,C col within tile

  // --- Stage B into LDS with one TDM descriptor (wave 0 only) -------------
  if (wave == 0) {
    const unsigned long long ga = (unsigned long long)(const void*)B;
    const unsigned ldsOff = (unsigned)(size_t)(void*)ldsW;  // LDS byte offset
    u32x4 g0 = {};
    g0[0] = 1u;                                   // count=1 (valid user D#)
    g0[1] = ldsOff;                               // lds_addr (bytes)
    g0[2] = (unsigned)(ga & 0xFFFFFFFFu);         // global_addr[31:0]
    g0[3] = (unsigned)((ga >> 32) & 0x1FFFFFFu)   // global_addr[56:32]
            | 0x80000000u;                        // type=2 ("image")
    i32x8 g1 = {};
    g1[0] = (int)(2u << 16);                      // data_size = 4 bytes
    g1[1] = (int)((unsigned)(ldn & 0xFFFF) << 16);   // tensor_dim0[15:0]
    g1[2] = (int)((unsigned)(K & 0xFFFF) << 16);     // dim0 hi=0 | tensor_dim1[15:0]
    g1[3] = (int)((unsigned)(ldn & 0xFFFF) << 16);   // dim1 hi=0 | tile_dim0
    g1[4] = (int)(unsigned)(K & 0xFFFF);             // tile_dim1 | tile_dim2=0
    g1[5] = ldn;                                     // tensor_dim0_stride[31:0]
    g1[6] = 0;                                       // stride hi / dim1_stride
    g1[7] = 0;
    i32x4 g2 = {}, g3 = {};                          // 2-D tensor: unused
#if defined(__clang_major__) && __clang_major__ >= 23
    i32x8 g4 = {};
    __builtin_amdgcn_tensor_load_to_lds(g0, g1, g2, g3, g4, 0);
#else
    __builtin_amdgcn_tensor_load_to_lds(g0, g1, g2, g3, 0);
#endif
    __builtin_amdgcn_s_wait_tensorcnt(0);
  }
  __syncthreads();

  // --- Tile assignment (tail waves exit after the barrier) ----------------
  const int CT = ldn >> 4;                    // column tiles (8 or 1)
  const int ct = wave % CT;
  const int rtile = blockIdx.x * (8 / CT) + wave / CT;
  if (rtile >= mtiles) return;
  const size_t row0 = (size_t)rtile * 16;
  const int    col0 = ct * 16;

  v8f acc = {};
  for (int k0 = 0; k0 < K; k0 += 32) {
    v16h a = {}, b = {};
#pragma unroll
    for (int r = 0; r < 8; ++r) {
      const int ka = k0 + (r < 4 ? 8 * g + 2 * r : 16 + 8 * g + 2 * (r - 4));
      const float2 pa = *(const float2*)(A + (row0 + mn) * (size_t)K + ka);
      a[2 * r]     = (_Float16)pa.x;
      a[2 * r + 1] = (_Float16)pa.y;
      const int kb = k0 + (r < 4 ? 8 * g + 2 * r : 16 + 8 * g + 2 * (r - 4));
      b[2 * r]     = (_Float16)ldsW[(size_t)kb * ldn + col0 + mn];
      b[2 * r + 1] = (_Float16)ldsW[(size_t)(kb + 1) * ldn + col0 + mn];
    }
    acc = __builtin_amdgcn_wmma_f32_16x16x32_f16(
        /*neg_a=*/false, a, /*neg_b=*/false, b,
        /*c_mod=*/(short)0, acc, /*reuse_a=*/false, /*reuse_b=*/false);
  }
#pragma unroll
  for (int r = 0; r < 8; ++r)
    C[(row0 + 8 * g + r) * (size_t)ldn + col0 + mn] = acc[r];
}

// ---------------------------------------------------------------------------
// Per-(node,head): attention logits al_src/al_dst; seed the segment max with
// the self-loop edge value (add_self_loops=True => every dst has >= 1 edge).
// ---------------------------------------------------------------------------
template <int H, int C>
__global__ void attn_node(const float* __restrict__ hfeat,   // [N, H*C]
                          const float* __restrict__ a_src,   // [H, C]
                          const float* __restrict__ a_dst,   // [H, C]
                          float* __restrict__ al_src,        // [N*H]
                          float* __restrict__ al_dst,        // [N*H]
                          float* __restrict__ m,             // [N*H] seg-max
                          int N) {
  const int i = blockIdx.x * blockDim.x + threadIdx.x;  // n*H + h
  if (i >= N * H) return;
  const int n = i / H, h = i % H;
  const float* row = hfeat + (size_t)n * (H * C) + h * C;
  float s = 0.f, d = 0.f;
#pragma unroll
  for (int c = 0; c < C; ++c) {
    s += row[c] * a_src[h * C + c];
    d += row[c] * a_dst[h * C + c];
  }
  al_src[i] = s;
  al_dst[i] = d;
  m[i] = leakyf(s + d);   // self-loop edge value
}

template <int H>
__global__ void edge_max(const int* __restrict__ src, const int* __restrict__ dst,
                         const float* __restrict__ al_src,
                         const float* __restrict__ al_dst,
                         float* __restrict__ m, int E) {
  const int e = blockIdx.x * blockDim.x + threadIdx.x;
  if (e >= E) return;
  const int s = src[e], d = dst[e];
#pragma unroll
  for (int h = 0; h < H; ++h)
    atomicMaxF(&m[d * H + h], leakyf(al_src[s * H + h] + al_dst[d * H + h]));
}

// Initialize denom with the self-loop contribution exp(self_e - m).
template <int H>
__global__ void node_denom(const float* __restrict__ al_src,
                           const float* __restrict__ al_dst,
                           const float* __restrict__ m,
                           float* __restrict__ denom, int NH) {
  const int i = blockIdx.x * blockDim.x + threadIdx.x;
  if (i >= NH) return;
  denom[i] = expf(leakyf(al_src[i] + al_dst[i]) - m[i]);
}

template <int H>
__global__ void edge_sum(const int* __restrict__ src, const int* __restrict__ dst,
                         const float* __restrict__ al_src,
                         const float* __restrict__ al_dst,
                         const float* __restrict__ m,
                         float* __restrict__ denom, int E) {
  const int idx = blockIdx.x * blockDim.x + threadIdx.x;
  if (idx >= E * H) return;
  const int e = idx / H, h = idx % H;
  const int s = src[e], d = dst[e];
  const float ev = leakyf(al_src[s * H + h] + al_dst[d * H + h]);
  atomicAdd(&denom[d * H + h], expf(ev - m[d * H + h]));
}

// agg[n] = h[n] * alpha_self  (exclusive writes; runs before edge_agg)
template <int H, int C>
__global__ void node_agg(const float* __restrict__ hfeat,
                         const float* __restrict__ al_src,
                         const float* __restrict__ al_dst,
                         const float* __restrict__ m,
                         const float* __restrict__ denom,
                         float* __restrict__ agg, int N) {
  const int i = blockIdx.x * blockDim.x + threadIdx.x;  // n*H + h
  if (i >= N * H) return;
  const int n = i / H, h = i % H;
  const float alpha =
      expf(leakyf(al_src[i] + al_dst[i]) - m[i]) / (denom[i] + 1e-16f);
  const float* hr = hfeat + (size_t)n * (H * C) + h * C;
  float* od = agg + (size_t)n * (H * C) + h * C;
#pragma unroll
  for (int c = 0; c < C; ++c) od[c] = hr[c] * alpha;
}

template <int H, int C>
__global__ void edge_agg(const int* __restrict__ src, const int* __restrict__ dst,
                         const float* __restrict__ hfeat,
                         const float* __restrict__ al_src,
                         const float* __restrict__ al_dst,
                         const float* __restrict__ m,
                         const float* __restrict__ denom,
                         float* __restrict__ agg, int E) {
  const int idx = blockIdx.x * blockDim.x + threadIdx.x;
  if (idx >= E * H) return;
  const int e = idx / H, h = idx % H;
  const int s = src[e], d = dst[e];
  const float ev = leakyf(al_src[s * H + h] + al_dst[d * H + h]);
  const float alpha = expf(ev - m[d * H + h]) / (denom[d * H + h] + 1e-16f);
  const float* hs = hfeat + (size_t)s * (H * C) + h * C;
  __builtin_prefetch(hs, 0, 0);  // global_prefetch_b8 of the gathered row
  float* od = agg + (size_t)d * (H * C) + h * C;
#pragma unroll
  for (int c = 0; c < C; ++c) atomicAdd(&od[c], hs[c] * alpha);
}

// In-place h2 = elu(agg1 + b1)
__global__ void elu_bias(float* __restrict__ v, const float* __restrict__ bias,
                         int total) {
  const int i = blockIdx.x * blockDim.x + threadIdx.x;
  if (i >= total) return;
  const float t = v[i] + bias[i % HC];
  v[i] = t > 0.f ? t : expm1f(t);
}

__global__ void final_logsoftmax(const float* __restrict__ agg2,
                                 const float* __restrict__ b2,
                                 float* __restrict__ out, int N) {
  const int n = blockIdx.x * blockDim.x + threadIdx.x;
  if (n >= N) return;
  float v[OUT_CH];
  float mx = -INFINITY;
#pragma unroll
  for (int c = 0; c < OUT_CH; ++c) {
    v[c] = agg2[(size_t)n * OUT_CH + c] + b2[c];
    mx = fmaxf(mx, v[c]);
  }
  float se = 0.f;
#pragma unroll
  for (int c = 0; c < OUT_CH; ++c) se += expf(v[c] - mx);
  const float ls = logf(se);
#pragma unroll
  for (int c = 0; c < OUT_CH; ++c) out[(size_t)n * OUT_CH + c] = v[c] - mx - ls;
}

static inline int gridFor(long long n, int b) { return (int)((n + b - 1) / b); }

extern "C" void kernel_launch(void* const* d_in, const int* in_sizes, int n_in,
                              void* d_out, int out_size, void* d_ws, size_t ws_size,
                              hipStream_t stream) {
  const float* x      = (const float*)d_in[0];
  const int*   ei     = (const int*)  d_in[1];
  /* d_in[2] edge_attr: ignored (PyG GATConv with edge_dim=None) */
  const float* W1     = (const float*)d_in[3];
  const float* a_src1 = (const float*)d_in[4];
  const float* a_dst1 = (const float*)d_in[5];
  const float* b1     = (const float*)d_in[6];
  const float* W2     = (const float*)d_in[7];
  const float* a_src2 = (const float*)d_in[8];
  const float* a_dst2 = (const float*)d_in[9];
  const float* b2     = (const float*)d_in[10];
  float* out = (float*)d_out;

  const int N = in_sizes[0] / IN_CH;   // 50000 (multiple of 16)
  const int E = in_sizes[1] / 2;       // 800000
  const int* src = ei;
  const int* dst = ei + E;

  // Workspace carve-out (~61.5 MB of f32)
  float* p    = (float*)d_ws;
  float* h1   = p; p += (size_t)N * HC;      // x @ W1
  float* agg1 = p; p += (size_t)N * HC;      // layer-1 aggregate -> h2 (in place)
  float* als1 = p; p += (size_t)N * HEADS;
  float* ald1 = p; p += (size_t)N * HEADS;
  float* m1   = p; p += (size_t)N * HEADS;
  float* dn1  = p; p += (size_t)N * HEADS;
  float* z2   = p; p += (size_t)N * OUT_CH;  // h2 @ W2
  float* agg2 = p; p += (size_t)N * OUT_CH;
  float* als2 = p; p += (size_t)N;
  float* ald2 = p; p += (size_t)N;
  float* m2   = p; p += (size_t)N;
  float* dn2  = p; p += (size_t)N;

  const int B = 256;
  const int mtiles = N / 16;  // 3125

  // ----- Layer 1 (H=4, C=32, concat) -----
  // 8 waves/block = 1 row-tile x 8 col-tiles; LDS = W1 (64 KB f32)
  gemm_wmma_f16<<<mtiles, 256, (size_t)IN_CH * HC * 4, stream>>>(
      x, W1, h1, IN_CH, HC, mtiles);
  attn_node<HEADS, HID><<<gridFor((long long)N * HEADS, B), B, 0, stream>>>(
      h1, a_src1, a_dst1, als1, ald1, m1, N);
  edge_max<HEADS><<<gridFor(E, B), B, 0, stream>>>(src, dst, als1, ald1, m1, E);
  node_denom<HEADS><<<gridFor((long long)N * HEADS, B), B, 0, stream>>>(
      als1, ald1, m1, dn1, N * HEADS);
  edge_sum<HEADS><<<gridFor((long long)E * HEADS, B), B, 0, stream>>>(
      src, dst, als1, ald1, m1, dn1, E);
  node_agg<HEADS, HID><<<gridFor((long long)N * HEADS, B), B, 0, stream>>>(
      h1, als1, ald1, m1, dn1, agg1, N);
  edge_agg<HEADS, HID><<<gridFor((long long)E * HEADS, B), B, 0, stream>>>(
      src, dst, h1, als1, ald1, m1, dn1, agg1, E);
  elu_bias<<<gridFor((long long)N * HC, B), B, 0, stream>>>(agg1, b1, N * HC);

  // ----- Layer 2 (H=1, C=16, mean over 1 head == identity) -----
  // 8 waves/block = 8 row-tiles x 1 col-tile; LDS = W2 (8 KB f32)
  gemm_wmma_f16<<<gridFor(mtiles, 8), 256, (size_t)HC * OUT_CH * 4, stream>>>(
      agg1, W2, z2, HC, OUT_CH, mtiles);
  attn_node<1, OUT_CH><<<gridFor(N, B), B, 0, stream>>>(
      z2, a_src2, a_dst2, als2, ald2, m2, N);
  edge_max<1><<<gridFor(E, B), B, 0, stream>>>(src, dst, als2, ald2, m2, E);
  node_denom<1><<<gridFor(N, B), B, 0, stream>>>(als2, ald2, m2, dn2, N);
  edge_sum<1><<<gridFor(E, B), B, 0, stream>>>(src, dst, als2, ald2, m2, dn2, E);
  node_agg<1, OUT_CH><<<gridFor(N, B), B, 0, stream>>>(
      z2, als2, ald2, m2, dn2, agg2, N);
  edge_agg<1, OUT_CH><<<gridFor(E, B), B, 0, stream>>>(
      src, dst, z2, als2, ald2, m2, dn2, agg2, E);
  final_logsoftmax<<<gridFor(N, B), B, 0, stream>>>(agg2, b2, out, N);
}
